// NodeBlock_86844238725704
// MI455X (gfx1250) — matrix operand order repaired and checked by reading, not compiled, not run on hardware
//
#include <hip/hip_runtime.h>
#include <hip/hip_bf16.h>
#include <math.h>

typedef __attribute__((ext_vector_type(16))) __bf16 v16bf;
typedef __attribute__((ext_vector_type(8)))  __bf16 v8bf;
typedef __attribute__((ext_vector_type(8)))  float  v8f;

#define BATCH 512
#define NNODE 256
#define NODE_D 256
#define FEAT_IN 512
#define INP_D 576          // 512 LN'd + 64 globals
#define KT_IN 18           // 576/32
#define FEAT_HID 1024
#define KT_HID 32          // 1024/32
#define FEAT_OUT 256
#define ATTN_HID 256
#define KT_AH 8            // 256/32
#define NROWS (BATCH*NNODE)   // 131072

// ---- workspace layout (bytes) ----
#define OFF_INPA   ((size_t)0)                         // bf16 [8192 rowtiles][18 kt][32 lane][16]
#define OFF_W1P    ((size_t)150994944)                 // bf16 64*18 frags * 1KB
#define OFF_W2P    ((size_t)152174592)                 // bf16 16*32 frags
#define OFF_AW1P   ((size_t)152698880)                 // bf16 16*18 frags
#define OFF_AW2P   ((size_t)152993792)                 // bf16 1*8 frags (N padded 4->16)
#define OFF_SCORES ((size_t)153001984)                 // f32 [NROWS][4]

#define WMMA_BF16(a, b, c) __builtin_amdgcn_wmma_f32_16x16x32_bf16( \
    false, (a), false, (b), (short)0, (c), false, false)

// A-fragment K index for 16-bit 16x32 tiles (ISA 7.12.2)
__device__ __forceinline__ int kmapA(int lane, int e) {
    return (e < 8 ? e : e + 8) + ((lane & 16) ? 8 : 0);
}
// B-fragment K index for 16-bit 32x16 tiles
__device__ __forceinline__ int kmapB(int lane, int e) {
    return (e < 8 ? e : e + 8) + ((lane & 16) ? 8 : 0);
}

__device__ __forceinline__ float wred_sum(float v) {
    for (int off = 16; off >= 1; off >>= 1) v += __shfl_xor(v, off, 32);
    return v;
}
__device__ __forceinline__ float wred_max(float v) {
    for (int off = 16; off >= 1; off >>= 1) v = fmaxf(v, __shfl_xor(v, off, 32));
    return v;
}

// -------------------------------------------------------------------------
// Kernel 1: LayerNorm + globals concat, emit bf16 input in WMMA-A layout
// -------------------------------------------------------------------------
__global__ __launch_bounds__(256) void k_prep(
    const float* __restrict__ nodes, const float* __restrict__ pe,
    const float* __restrict__ globs, const float* __restrict__ ln_g,
    const float* __restrict__ ln_b, __bf16* __restrict__ inpA) {
    __shared__ float xrow[16][FEAT_IN];
    __shared__ float muS[16], rsS[16];
    const int blk = blockIdx.x;          // row tile (16 rows)
    const int m0  = blk * 16;
    const int t = threadIdx.x, w = t >> 5, lane = t & 31;

    for (int rr = w * 2; rr < w * 2 + 2; ++rr) {
        const int m = m0 + rr;
        float s = 0.f, s2 = 0.f;
        #pragma unroll
        for (int j = 0; j < 16; ++j) {
            int k = j * 32 + lane;
            float v = (k < NODE_D) ? nodes[(size_t)m * NODE_D + k]
                                   : pe[(size_t)m * NODE_D + (k - NODE_D)];
            xrow[rr][k] = v;
            s += v; s2 += v * v;
        }
        s = wred_sum(s); s2 = wred_sum(s2);
        if (lane == 0) {
            float mu = s * (1.f / FEAT_IN);
            float var = s2 * (1.f / FEAT_IN) - mu * mu;
            muS[rr] = mu;
            rsS[rr] = rsqrtf(var + 1e-5f);
        }
    }
    __syncthreads();

    const int b = m0 >> 8;               // tiles never straddle batches
    for (int kt = t >> 5; kt < KT_IN; kt += 8) {
        const int rr = lane & 15;
        const float mu = muS[rr], rs = rsS[rr];
        v16bf o;
        #pragma unroll
        for (int e = 0; e < 16; ++e) {
            int k = kt * 32 + kmapA(lane, e);
            float v;
            if (k < FEAT_IN) v = (xrow[rr][k] - mu) * rs * ln_g[k] + ln_b[k];
            else             v = globs[(size_t)b * 64 + (k - FEAT_IN)];
            o[e] = (__bf16)v;
        }
        *(v16bf*)(inpA + (((size_t)blk * KT_IN + kt) * 32 + lane) * 16) = o;
    }
}

// -------------------------------------------------------------------------
// Kernel 2: pack fp32 weights [K,N] into bf16 WMMA-B fragments
// -------------------------------------------------------------------------
__global__ __launch_bounds__(256) void k_packw(
    const float* __restrict__ W, int Ndim, int KT, int nfrag,
    __bf16* __restrict__ out) {
    int id = blockIdx.x * 256 + threadIdx.x;
    int frag = id >> 5, lane = id & 31;
    if (frag >= nfrag) return;
    int nt = frag / KT, kt = frag % KT;
    int n = nt * 16 + (lane & 15);
    v16bf o;
    #pragma unroll
    for (int e = 0; e < 16; ++e) {
        int k = kt * 32 + kmapB(lane, e);
        float v = (n < Ndim) ? W[(size_t)k * Ndim + n] : 0.f;
        o[e] = (__bf16)v;
    }
    *(v16bf*)(out + (size_t)frag * 512 + lane * 16) = o;
}

// -------------------------------------------------------------------------
// Kernel 3: fused feature MLP (576->1024 SiLU ->256) + mask + residual
// 32 rows per block, 512 threads (16 waves), hidden bf16 in LDS (64 KB)
// -------------------------------------------------------------------------
__global__ __launch_bounds__(512) void k_feat(
    const __bf16* __restrict__ inpA, const __bf16* __restrict__ w1p,
    const float* __restrict__ fb1, const __bf16* __restrict__ w2p,
    const float* __restrict__ fb2, const float* __restrict__ nodes,
    const int* __restrict__ mask, float* __restrict__ out_nodes) {
    __shared__ __bf16 hbuf[32 * FEAT_HID];   // 64 KB, row-major [32][1024]
    const int blk = blockIdx.x;              // 32-row block
    const int t = threadIdx.x, w = t >> 5, lane = t & 31;
    const int r = w & 1;
    const int nt0 = (w >> 1) * 8;            // 8 hidden n-tiles per wave (stage 1)
    const int RT = blk * 2 + r;
    const int ncol = lane & 15;
    const int mhalf = (lane >> 4) & 1;

    auto ldA = [&](int kt) -> v16bf {
        return *(const v16bf*)(inpA + (((size_t)RT * KT_IN + kt) * 32 + lane) * 16);
    };
    auto ldB1 = [&](int j, int kt) -> v16bf {
        return *(const v16bf*)(w1p + ((size_t)((nt0 + j) * KT_IN + kt)) * 512 + lane * 16);
    };

    // ---- stage 1: h = silu(inp @ W1 + b1), 8 accum tiles per wave ----
    v8f acc[8];
    #pragma unroll
    for (int j = 0; j < 8; ++j)
        #pragma unroll
        for (int p = 0; p < 8; ++p) acc[j][p] = 0.f;

    v16bf a_cur = ldA(0), a_nxt = a_cur;
    v16bf rb[4];
    #pragma unroll
    for (int q = 0; q < 4; ++q) rb[q] = ldB1(q, 0);

    for (int kt = 0; kt < KT_IN; ++kt) {
        #pragma unroll
        for (int jg = 0; jg < 2; ++jg) {
            v16bf t0 = rb[0], t1 = rb[1], t2 = rb[2], t3 = rb[3];
            if (jg == 0 && kt + 1 < KT_IN) a_nxt = ldA(kt + 1);
            const int nkt = (jg == 1) ? kt + 1 : kt;
            const int nj  = (jg == 1) ? 0 : 4;
            if (nkt < KT_IN) {
                rb[0] = ldB1(nj + 0, nkt); rb[1] = ldB1(nj + 1, nkt);
                rb[2] = ldB1(nj + 2, nkt); rb[3] = ldB1(nj + 3, nkt);
            }
            const int j0 = jg * 4;
            acc[j0 + 0] = WMMA_BF16(a_cur, t0, acc[j0 + 0]);
            acc[j0 + 1] = WMMA_BF16(a_cur, t1, acc[j0 + 1]);
            acc[j0 + 2] = WMMA_BF16(a_cur, t2, acc[j0 + 2]);
            acc[j0 + 3] = WMMA_BF16(a_cur, t3, acc[j0 + 3]);
        }
        a_cur = a_nxt;
    }

    #pragma unroll
    for (int j = 0; j < 8; ++j) {
        int n = (nt0 + j) * 16 + ncol;
        float bias = fb1[n];
        #pragma unroll
        for (int p = 0; p < 8; ++p) {
            float x = acc[j][p] + bias;
            float h = x / (1.f + __expf(-x));           // SiLU
            hbuf[(r * 16 + mhalf * 8 + p) * FEAT_HID + n] = (__bf16)h;
        }
    }
    __syncthreads();

    // ---- stage 2: feats = h @ W2 + b2 ; mask ; + nodes ----
    const int nt2 = (w >> 1) * 2;            // 2 output n-tiles per wave
    const int mloc = r * 16 + (lane & 15);
    auto ldH = [&](int kt) -> v16bf {
        const __bf16* base = &hbuf[mloc * FEAT_HID + kt * 32 + ((lane & 16) ? 8 : 0)];
        v8bf lo = *(const v8bf*)(base);
        v8bf hi = *(const v8bf*)(base + 16);
        v16bf a;
        #pragma unroll
        for (int e = 0; e < 8; ++e) { a[e] = lo[e]; a[e + 8] = hi[e]; }
        return a;
    };
    auto ldB2 = [&](int j, int kt) -> v16bf {
        return *(const v16bf*)(w2p + ((size_t)((nt2 + j) * KT_HID + kt)) * 512 + lane * 16);
    };

    v8f acc2[2];
    #pragma unroll
    for (int j = 0; j < 2; ++j)
        #pragma unroll
        for (int p = 0; p < 8; ++p) acc2[j][p] = 0.f;

    v16bf h_cur = ldH(0), h_nxt = h_cur;
    v16bf sb[2];
    sb[0] = ldB2(0, 0); sb[1] = ldB2(1, 0);

    for (int kt = 0; kt < KT_HID; ++kt) {
        v16bf t0 = sb[0], t1 = sb[1];
        if (kt + 1 < KT_HID) {
            h_nxt = ldH(kt + 1);
            sb[0] = ldB2(0, kt + 1); sb[1] = ldB2(1, kt + 1);
        }
        acc2[0] = WMMA_BF16(h_cur, t0, acc2[0]);
        acc2[1] = WMMA_BF16(h_cur, t1, acc2[1]);
        h_cur = h_nxt;
    }

    const int mrow0 = blk * 32 + r * 16 + mhalf * 8;
    #pragma unroll
    for (int j = 0; j < 2; ++j) {
        int n = (nt2 + j) * 16 + ncol;
        float bias = fb2[n];
        #pragma unroll
        for (int p = 0; p < 8; ++p) {
            int m = mrow0 + p;
            float f = acc2[j][p] + bias;
            f = mask[m] ? f : 0.f;
            out_nodes[(size_t)m * FEAT_OUT + n] = f + nodes[(size_t)m * NODE_D + n];
        }
    }
}

// -------------------------------------------------------------------------
// Kernel 4: fused attention MLP (576->256 SiLU ->4, padded to 16) -> scores/16
// 512 threads (16 waves)
// -------------------------------------------------------------------------
__global__ __launch_bounds__(512) void k_attn(
    const __bf16* __restrict__ inpA, const __bf16* __restrict__ aw1p,
    const float* __restrict__ ab1, const __bf16* __restrict__ aw2p,
    const float* __restrict__ ab2, float* __restrict__ scores) {
    __shared__ __bf16 hbuf[32 * ATTN_HID];   // 16 KB
    const int blk = blockIdx.x;
    const int t = threadIdx.x, w = t >> 5, lane = t & 31;
    const int r = w & 1;
    const int nt0 = (w >> 1) * 2;            // 2 hidden n-tiles per wave
    const int RT = blk * 2 + r;
    const int ncol = lane & 15;
    const int mhalf = (lane >> 4) & 1;

    auto ldA = [&](int kt) -> v16bf {
        return *(const v16bf*)(inpA + (((size_t)RT * KT_IN + kt) * 32 + lane) * 16);
    };
    auto ldB1 = [&](int j, int kt) -> v16bf {
        return *(const v16bf*)(aw1p + ((size_t)((nt0 + j) * KT_IN + kt)) * 512 + lane * 16);
    };

    v8f acc[2];
    #pragma unroll
    for (int j = 0; j < 2; ++j)
        #pragma unroll
        for (int p = 0; p < 8; ++p) acc[j][p] = 0.f;

    v16bf a_cur = ldA(0), a_nxt = a_cur;
    v16bf rb[2];
    rb[0] = ldB1(0, 0); rb[1] = ldB1(1, 0);

    for (int kt = 0; kt < KT_IN; ++kt) {
        v16bf t0 = rb[0], t1 = rb[1];
        if (kt + 1 < KT_IN) {
            a_nxt = ldA(kt + 1);
            rb[0] = ldB1(0, kt + 1); rb[1] = ldB1(1, kt + 1);
        }
        acc[0] = WMMA_BF16(a_cur, t0, acc[0]);
        acc[1] = WMMA_BF16(a_cur, t1, acc[1]);
        a_cur = a_nxt;
    }

    #pragma unroll
    for (int j = 0; j < 2; ++j) {
        int n = (nt0 + j) * 16 + ncol;
        float bias = ab1[n];
        #pragma unroll
        for (int p = 0; p < 8; ++p) {
            float x = acc[j][p] + bias;
            float h = x / (1.f + __expf(-x));
            hbuf[(r * 16 + mhalf * 8 + p) * ATTN_HID + n] = (__bf16)h;
        }
    }
    __syncthreads();

    if (w < 2) {   // waves 0,1 compute the two 16x16 score tiles (only 4 cols live)
        const int r2 = w;
        const int mloc = r2 * 16 + (lane & 15);
        auto ldH = [&](int kt) -> v16bf {
            const __bf16* base = &hbuf[mloc * ATTN_HID + kt * 32 + ((lane & 16) ? 8 : 0)];
            v8bf lo = *(const v8bf*)(base);
            v8bf hi = *(const v8bf*)(base + 16);
            v16bf a;
            #pragma unroll
            for (int e = 0; e < 8; ++e) { a[e] = lo[e]; a[e + 8] = hi[e]; }
            return a;
        };
        v8f acc2;
        #pragma unroll
        for (int p = 0; p < 8; ++p) acc2[p] = 0.f;

        v16bf h_cur = ldH(0), h_nxt = h_cur;
        v16bf b_cur = *(const v16bf*)(aw2p + lane * 16), b_nxt = b_cur;
        for (int kt = 0; kt < KT_AH; ++kt) {
            if (kt + 1 < KT_AH) {
                h_nxt = ldH(kt + 1);
                b_nxt = *(const v16bf*)(aw2p + (size_t)(kt + 1) * 512 + lane * 16);
            }
            acc2 = WMMA_BF16(h_cur, b_cur, acc2);
            h_cur = h_nxt; b_cur = b_nxt;
        }
        if (ncol < 4) {
            const int mrow0 = blk * 32 + r2 * 16 + mhalf * 8;
            float bias = ab2[ncol];
            #pragma unroll
            for (int p = 0; p < 8; ++p)
                scores[(size_t)(mrow0 + p) * 4 + ncol] = (acc2[p] + bias) * 0.0625f; // /sqrt(256)
        }
    }
}

// -------------------------------------------------------------------------
// Kernel 5: masked softmax over nodes + attention pooling; one block per batch
// -------------------------------------------------------------------------
__global__ __launch_bounds__(256) void k_pool(
    const float* __restrict__ scores, const int* __restrict__ mask,
    const float* __restrict__ new_nodes, float* __restrict__ pooled) {
    __shared__ float wLds[NNODE * 4];
    const int b = blockIdx.x;
    const int t = threadIdx.x, w = t >> 5, lane = t & 31;

    if (w < 4) {                 // wave w handles head w
        const int h = w;
        float s[8]; float mx = -INFINITY;
        #pragma unroll
        for (int i = 0; i < 8; ++i) {
            int m = b * NNODE + i * 32 + lane;
            float v = mask[m] ? scores[(size_t)m * 4 + h] : -INFINITY;
            s[i] = v; mx = fmaxf(mx, v);
        }
        mx = wred_max(mx);
        float sum = 0.f;
        #pragma unroll
        for (int i = 0; i < 8; ++i) {
            float e = (s[i] == -INFINITY) ? 0.f : __expf(s[i] - mx);
            s[i] = e; sum += e;
        }
        sum = wred_sum(sum);
        float inv = (sum > 0.f) ? 1.f / sum : 0.f;   // nan_to_num for all-masked
        #pragma unroll
        for (int i = 0; i < 8; ++i) wLds[(i * 32 + lane) * 4 + h] = s[i] * inv;
    }
    __syncthreads();

    const int d = t, h = d >> 6;
    float acc = 0.f;
    #pragma unroll 4
    for (int node = 0; node < NNODE; ++node)
        acc += new_nodes[((size_t)b * NNODE + node) * FEAT_OUT + d] * wLds[node * 4 + h];
    pooled[(size_t)b * FEAT_OUT + d] = acc;
}

// -------------------------------------------------------------------------
extern "C" void kernel_launch(void* const* d_in, const int* in_sizes, int n_in,
                              void* d_out, int out_size, void* d_ws, size_t ws_size,
                              hipStream_t stream) {
    const float* nodes = (const float*)d_in[0];
    const float* pe    = (const float*)d_in[1];
    const int*   mask  = (const int*)d_in[2];
    const float* globs = (const float*)d_in[3];
    const float* ln_g  = (const float*)d_in[4];
    const float* ln_b  = (const float*)d_in[5];
    const float* fw1   = (const float*)d_in[6];
    const float* fb1   = (const float*)d_in[7];
    const float* fw2   = (const float*)d_in[8];
    const float* fb2   = (const float*)d_in[9];
    const float* aw1   = (const float*)d_in[10];
    const float* ab1   = (const float*)d_in[11];
    const float* aw2   = (const float*)d_in[12];
    const float* ab2   = (const float*)d_in[13];

    char* ws = (char*)d_ws;
    __bf16* inpA   = (__bf16*)(ws + OFF_INPA);
    __bf16* w1p    = (__bf16*)(ws + OFF_W1P);
    __bf16* w2p    = (__bf16*)(ws + OFF_W2P);
    __bf16* aw1p   = (__bf16*)(ws + OFF_AW1P);
    __bf16* aw2p   = (__bf16*)(ws + OFF_AW2P);
    float*  scores = (float*)(ws + OFF_SCORES);

    float* out_nodes = (float*)d_out;                         // [B*N, 256]
    float* pooled    = out_nodes + (size_t)NROWS * FEAT_OUT;  // [B, 256]

    // 1) LN + pack input (8192 row tiles)
    k_prep<<<NROWS / 16, 256, 0, stream>>>(nodes, pe, globs, ln_g, ln_b, inpA);

    // 2) pack weights (nfrag = NT*KT)
    k_packw<<<(64 * KT_IN * 32 + 255) / 256, 256, 0, stream>>>(fw1, FEAT_HID, KT_IN, 64 * KT_IN, w1p);
    k_packw<<<(16 * KT_HID * 32 + 255) / 256, 256, 0, stream>>>(fw2, FEAT_OUT, KT_HID, 16 * KT_HID, w2p);
    k_packw<<<(16 * KT_IN * 32 + 255) / 256, 256, 0, stream>>>(aw1, ATTN_HID, KT_IN, 16 * KT_IN, aw1p);
    k_packw<<<(1 * KT_AH * 32 + 255) / 256, 256, 0, stream>>>(aw2, 4, KT_AH, 1 * KT_AH, aw2p);

    // 3) fused feature MLP -> new_nodes  (4096 blocks of 32 rows, 512 thr)
    k_feat<<<NROWS / 32, 512, 0, stream>>>(inpA, w1p, fb1, w2p, fb2, nodes, mask, out_nodes);

    // 4) fused attention MLP -> scaled scores
    k_attn<<<NROWS / 32, 512, 0, stream>>>(inpA, aw1p, ab1, aw2p, ab2, scores);

    // 5) masked softmax + attention pooling -> pooled
    k_pool<<<BATCH, 256, 0, stream>>>(scores, mask, out_nodes, pooled);
}
